// LinearAttention_2748779070113
// MI455X (gfx1250) — matrix operand order repaired and checked by reading, not compiled, not run on hardware
//
#include <hip/hip_runtime.h>

typedef __attribute__((ext_vector_type(16))) __bf16 v16bf;
typedef __attribute__((ext_vector_type(8)))  __bf16 bf16x8;
typedef __attribute__((ext_vector_type(4)))  __bf16 bf16x4;
typedef __attribute__((ext_vector_type(8)))  float  v8f;
typedef __attribute__((ext_vector_type(4)))  unsigned int u32x4;
typedef __attribute__((ext_vector_type(8)))  unsigned int u32x8;

static constexpr int BM = 128;
static constexpr int BN = 128;
static constexpr int BK = 32;                 // one wmma K-step
static constexpr int LDSS = BK + 8;           // 40 halves = 80B row (64B data + 16B TDM pad)

// Issue one TDM DMA: 128 rows x 32 bf16 tile (row stride ld elems) -> LDS at lds_off,
// padded 16B per 64B row so LDS row stride == LDSS halves. Uniform args -> SGPR D#.
__device__ __forceinline__ void tdm_load_tile(unsigned lds_off, const __bf16* gptr, int ld_elems)
{
    u32x4 g0;
    unsigned long long ga = (unsigned long long)gptr;
    g0[0] = 1u;                                   // count=1, user mode
    g0[1] = lds_off;                              // lds_addr (bytes)
    g0[2] = (unsigned)ga;                         // global_addr[31:0]
    g0[3] = (unsigned)(ga >> 32) | 0x80000000u;   // global_addr[56:32] | type=2 (bits 127:126)

    u32x8 g1;
    g1[0] = (1u << 16)        // data_size = 2 bytes
          | (1u << 20)        // pad_enable
          | (3u << 22)        // pad_interval: 16 DWORDs (=64B row)
          | (3u << 25);       // pad_amount: 4 DWORDs (=16B)
    g1[1] = (unsigned)(BK & 0xffff) << 16;            // tensor_dim0[15:0] = 32
    g1[2] = ((unsigned)(BM & 0xffff) << 16);          // dim0 hi=0 | tensor_dim1[15:0] = 128
    g1[3] = ((unsigned)BK << 16);                     // dim1 hi=0 | tile_dim0 = 32
    g1[4] = (unsigned)BM;                             // tile_dim1 = 128 | tile_dim2 = 0
    g1[5] = (unsigned)ld_elems;                       // tensor_dim0_stride[31:0]
    g1[6] = 0u;                                       // stride hi | dim1_stride lo (unused, 2D)
    g1[7] = 0u;

    u32x4 gz = {0u, 0u, 0u, 0u};                      // groups 2/3: dims 2..4 unused
    asm volatile("tensor_load_to_lds %0, %1, %2, %3"
                 :: "s"(g0), "s"(g1), "s"(gz), "s"(gz)
                 : "memory");
}

// C[M,N] = scale * (A[M,K] x Bt[N,K]^T) + bias[N]
// A row-major (lda), Bt is the N-major ("pre-transposed") operand (ldb).
// OUT_F32: fp32 output, else bf16. TRANS_STORE: store C transposed (C[n][m], ldc along m).
template<bool OUT_F32, bool TRANS_STORE>
__global__ __launch_bounds__(256)
void gemm_bf16_wmma(const __bf16* __restrict__ A, long aBatch, int lda,
                    const __bf16* __restrict__ Bt, long bBatch, int ldb,
                    void* __restrict__ Cp, long cBatch, int ldc,
                    const float* __restrict__ bias, float scale, int Kdim)
{
    __shared__ __bf16 As[2][BM * LDSS];
    __shared__ __bf16 Bs[2][BN * LDSS];

    const int tid  = threadIdx.x;
    const int lane = tid & 31;
    const int wave = tid >> 5;
    const int half = lane >> 4;       // 0/1: K sub-range select per ISA fragment layout
    const int l16  = lane & 15;
    const int wm   = wave >> 2;       // 0..1 -> 64 rows
    const int wn   = wave & 3;        // 0..3 -> 32 cols
    const long bm0 = (long)blockIdx.x * BM;
    const long bn0 = (long)blockIdx.y * BN;

    // Per-block tile bases (uniform).
    const __bf16* Atile = A  + (long)blockIdx.z * aBatch + bm0 * (long)lda;
    const __bf16* Btile = Bt + (long)blockIdx.z * bBatch + bn0 * (long)ldb;
    const unsigned ldsA0 = (unsigned)(unsigned long long)&As[0][0];
    const unsigned ldsA1 = (unsigned)(unsigned long long)&As[1][0];
    const unsigned ldsB0 = (unsigned)(unsigned long long)&Bs[0][0];
    const unsigned ldsB1 = (unsigned)(unsigned long long)&Bs[1][0];

    v8f acc[4][2] = {};

    const int nk = Kdim / BK;

    // Prologue: DMA stage 0 into buffer 0 (wave 0 drives the TDM).
    if (wave == 0) {
        tdm_load_tile(ldsA0, Atile, lda);
        tdm_load_tile(ldsB0, Btile, ldb);
    }

    for (int i = 0; i < nk; ++i) {
        const int buf = i & 1;
        if (wave == 0) {
            if (i + 1 < nk) {
                const int k1 = (i + 1) * BK;
                tdm_load_tile(buf ? ldsA0 : ldsA1, Atile + k1, lda);
                tdm_load_tile(buf ? ldsB0 : ldsB1, Btile + k1, ldb);
                __builtin_amdgcn_s_wait_tensorcnt(2);  // stage i landed; i+1 in flight
            } else {
                __builtin_amdgcn_s_wait_tensorcnt(0);  // final stage landed
            }
        }
        __syncthreads();   // publish stage i to all waves

        const __bf16* as = As[buf];
        const __bf16* bs = Bs[buf];

        // A fragment (16x32): half 0 -> K 0-7 & 16-23, half 1 -> K 8-15 & 24-31
        v16bf af[4];
#pragma unroll
        for (int mi = 0; mi < 4; ++mi) {
            const int r = wm * 64 + mi * 16 + l16;
            bf16x8 lo = *(const bf16x8*)&as[r * LDSS + half * 8];
            bf16x8 hi = *(const bf16x8*)&as[r * LDSS + 16 + half * 8];
            af[mi] = __builtin_shufflevector(lo, hi, 0,1,2,3,4,5,6,7,8,9,10,11,12,13,14,15);
        }
        // B fragment (32x16): lane l16 = column, half 0 -> K 0-7 & 16-23, half 1 -> K 8-15 & 24-31
        v16bf bfr[2];
#pragma unroll
        for (int ni = 0; ni < 2; ++ni) {
            const int r = wn * 32 + ni * 16 + l16;
            bf16x8 lo = *(const bf16x8*)&bs[r * LDSS + half * 8];
            bf16x8 hi = *(const bf16x8*)&bs[r * LDSS + 16 + half * 8];
            bfr[ni] = __builtin_shufflevector(lo, hi, 0,1,2,3,4,5,6,7,8,9,10,11,12,13,14,15);
        }

#pragma unroll
        for (int mi = 0; mi < 4; ++mi)
#pragma unroll
            for (int ni = 0; ni < 2; ++ni)
                acc[mi][ni] = __builtin_amdgcn_wmma_f32_16x16x32_bf16(
                    false, af[mi], false, bfr[ni], (short)0, acc[mi][ni], false, false);

        __syncthreads();   // all reads of buf done before stage i+2 overwrites it
    }

    // Store D fragments: VGPR r, lanes<16 -> M=r, lanes>=16 -> M=8+r; N = l16.
#pragma unroll
    for (int mi = 0; mi < 4; ++mi) {
#pragma unroll
        for (int ni = 0; ni < 2; ++ni) {
            const long col  = bn0 + wn * 32 + ni * 16 + l16;
            const long row0 = bm0 + wm * 64 + mi * 16 + half * 8;
            const float badd = bias ? bias[col] : 0.0f;
            if (TRANS_STORE) {
                __bf16* C = (__bf16*)Cp + (long)blockIdx.z * cBatch;
                bf16x8 pk;
#pragma unroll
                for (int r = 0; r < 8; ++r)
                    pk[r] = (__bf16)(acc[mi][ni][r] * scale + badd);
                *(bf16x8*)&C[col * (long)ldc + row0] = pk;   // C[n][m], 8 contiguous m
            } else if (OUT_F32) {
                float* C = (float*)Cp + (long)blockIdx.z * cBatch;
#pragma unroll
                for (int r = 0; r < 8; ++r)
                    C[(row0 + r) * (long)ldc + col] = acc[mi][ni][r] * scale + badd;
            } else {
                __bf16* C = (__bf16*)Cp + (long)blockIdx.z * cBatch;
#pragma unroll
                for (int r = 0; r < 8; ++r)
                    C[(row0 + r) * (long)ldc + col] = (__bf16)(acc[mi][ni][r] * scale + badd);
            }
        }
    }
}

// In-place masked softmax over rows of length S==2048 (bf16 storage, fp32 math).
__global__ __launch_bounds__(256)
void softmax_mask_kernel(__bf16* __restrict__ P, const int* __restrict__ mask, int S)
{
    const long row = blockIdx.x;            // b*S + q
    const int  b   = (int)(row / S);
    __bf16* p = P + row * (long)S;
    const int* mk = mask + (long)b * S;
    const int tid = threadIdx.x;
    __shared__ float red[256];

    float v[8];
    float mx = -3.0e38f;
#pragma unroll
    for (int i = 0; i < 8; ++i) {
        const int k = tid + (i << 8);
        float s = (float)p[k];
        if (mk[k] == 0) s = -1.0e9f;
        v[i] = s;
        mx = fmaxf(mx, s);
    }
    red[tid] = mx; __syncthreads();
    for (int off = 128; off > 0; off >>= 1) {
        if (tid < off) red[tid] = fmaxf(red[tid], red[tid + off]);
        __syncthreads();
    }
    const float rmax = red[0];
    __syncthreads();

    float sum = 0.f;
#pragma unroll
    for (int i = 0; i < 8; ++i) { v[i] = __expf(v[i] - rmax); sum += v[i]; }
    red[tid] = sum; __syncthreads();
    for (int off = 128; off > 0; off >>= 1) {
        if (tid < off) red[tid] += red[tid + off];
        __syncthreads();
    }
    const float inv = 1.0f / red[0];
#pragma unroll
    for (int i = 0; i < 8; ++i) {
        const int k = tid + (i << 8);
        p[k] = (__bf16)(v[i] * inv);
    }
}

__global__ __launch_bounds__(256)
void convert_f32_to_bf16(const float* __restrict__ x, __bf16* __restrict__ y, long n4)
{
    const long i = (long)blockIdx.x * 256 + threadIdx.x;
    if (i >= n4) return;
    const float4 v = ((const float4*)x)[i];
    bf16x4 o = { (__bf16)v.x, (__bf16)v.y, (__bf16)v.z, (__bf16)v.w };
    ((bf16x4*)y)[i] = o;
}

// Wt[n][k] = (bf16) W[k][n], LDS tile transpose, coalesced both directions.
__global__ __launch_bounds__(256)
void transpose_w_to_bf16(const float* __restrict__ W, __bf16* __restrict__ Wt, int D)
{
    __shared__ float tile[32][33];
    const int n0 = blockIdx.x * 32;
    const int k0 = blockIdx.y * 32;
    const int tx = threadIdx.x & 31;
    const int ty = threadIdx.x >> 5;   // 0..7
#pragma unroll
    for (int i = 0; i < 4; ++i) {
        const int r = ty + i * 8;
        tile[r][tx] = W[(long)(k0 + r) * D + (n0 + tx)];
    }
    __syncthreads();
#pragma unroll
    for (int i = 0; i < 4; ++i) {
        const int r = ty + i * 8;
        Wt[(long)(n0 + r) * D + (k0 + tx)] = (__bf16)tile[tx][r];
    }
}

extern "C" void kernel_launch(void* const* d_in, const int* in_sizes, int n_in,
                              void* d_out, int out_size, void* d_ws, size_t ws_size,
                              hipStream_t stream)
{
    (void)in_sizes; (void)n_in; (void)out_size; (void)ws_size;
    constexpr int  Bn = 4;
    constexpr int  S  = 2048;
    constexpr int  D  = 1024;
    constexpr long SD = (long)S * D;
    constexpr long SS = (long)S * S;

    const float* x    = (const float*)d_in[0];
    const int*   mask = (const int*)  d_in[1];
    const float* Wq   = (const float*)d_in[2];
    const float* bq   = (const float*)d_in[3];
    const float* Wk   = (const float*)d_in[4];
    const float* bk   = (const float*)d_in[5];
    const float* Wv   = (const float*)d_in[6];
    const float* bv   = (const float*)d_in[7];
    const float* Wo   = (const float*)d_in[8];
    const float* bo   = (const float*)d_in[9];
    float* out = (float*)d_out;

    char*  ws  = (char*)d_ws;
    size_t off = 0;
    auto take = [&](size_t bytes) -> char* {
        char* p = ws + off;
        off = (off + bytes + 255) & ~(size_t)255;
        return p;
    };
    __bf16* xb  = (__bf16*)take((size_t)Bn * SD * 2);   // 16 MB
    __bf16* Wtq = (__bf16*)take((size_t)D * D * 2);     // 2 MB each
    __bf16* Wtk = (__bf16*)take((size_t)D * D * 2);
    __bf16* Wtv = (__bf16*)take((size_t)D * D * 2);
    __bf16* Wto = (__bf16*)take((size_t)D * D * 2);
    __bf16* Qb  = (__bf16*)take((size_t)Bn * SD * 2);   // 16 MB
    __bf16* Kb  = (__bf16*)take((size_t)Bn * SD * 2);   // 16 MB
    __bf16* Vt  = (__bf16*)take((size_t)Bn * SD * 2);   // 16 MB (per batch [D][S])
    __bf16* P   = (__bf16*)take((size_t)Bn * SS * 2);   // 32 MB
    __bf16* ctx = (__bf16*)take((size_t)Bn * SD * 2);   // 16 MB

    // 1) x -> bf16
    const long n4 = (long)Bn * SD / 4;
    convert_f32_to_bf16<<<dim3((unsigned)(n4 / 256)), 256, 0, stream>>>(x, xb, n4);

    // 2) weights -> bf16, transposed to N-major
    dim3 tg(D / 32, D / 32);
    transpose_w_to_bf16<<<tg, 256, 0, stream>>>(Wq, Wtq, D);
    transpose_w_to_bf16<<<tg, 256, 0, stream>>>(Wk, Wtk, D);
    transpose_w_to_bf16<<<tg, 256, 0, stream>>>(Wv, Wtv, D);
    transpose_w_to_bf16<<<tg, 256, 0, stream>>>(Wo, Wto, D);

    // 3) Q, K (row-major), V (stored transposed per batch)
    dim3 gProj(S / BM, D / BN, Bn);
    gemm_bf16_wmma<false, false><<<gProj, 256, 0, stream>>>(
        xb, SD, D, Wtq, 0, D, Qb, SD, D, bq, 1.0f, D);
    gemm_bf16_wmma<false, false><<<gProj, 256, 0, stream>>>(
        xb, SD, D, Wtk, 0, D, Kb, SD, D, bk, 1.0f, D);
    gemm_bf16_wmma<false, true><<<gProj, 256, 0, stream>>>(
        xb, SD, D, Wtv, 0, D, Vt, SD, S, bv, 1.0f, D);

    // 4) scores = (Q . K^T) / sqrt(D)   (K row-major over D == N-major operand)
    dim3 gScore(S / BM, S / BN, Bn);
    gemm_bf16_wmma<false, false><<<gScore, 256, 0, stream>>>(
        Qb, SD, D, Kb, SD, D, P, SS, S, nullptr, 0.03125f, D);

    // 5) masked softmax, in place
    softmax_mask_kernel<<<dim3(Bn * S), 256, 0, stream>>>(P, mask, S);

    // 6) ctx = P . V   (Vt is the N-major operand, K-dim = S)
    gemm_bf16_wmma<false, false><<<gProj, 256, 0, stream>>>(
        P, SS, S, Vt, SD, S, ctx, SD, D, nullptr, 1.0f, S);

    // 7) out = ctx . Wo + bo  (fp32 output)
    gemm_bf16_wmma<true, false><<<gProj, 256, 0, stream>>>(
        ctx, SD, D, Wto, 0, D, out, SD, D, bo, 1.0f, D);
}
